// MorletConv_68848325755161
// MI455X (gfx1250) — compile-verified, matrix operand order checked by reading, not run on hardware
//
#include <hip/hip_runtime.h>

typedef _Float16 half_t;
typedef __attribute__((ext_vector_type(16))) _Float16 v16h;
typedef __attribute__((ext_vector_type(8)))  float    v8f;

#define TTOT   65536
#define CHANS  64
#define WLEN   25
#define ETAS   25
#define LOUT   (TTOT - WLEN + 1)   // 65512
#define WTIME  0.36f
#define TWO_PI 6.28318530717958647692f
#define ITER   8
#define WAVES_PER_BLOCK 8

// out[l,e,c] = sum_k W[e,k] * X[l+k,c]
// Per l: D(16x16) = A(16x32 f16) x B(32x16 f16); 2 M-tiles x 4 N-tiles.
// A columns k>=25 are zero, so B needs no predication, only legal addresses.
// Signed 32-bit element indices (no-wrap assumption) let the backend fold all
// compile-time offsets into imm24 and use saddr + 32-bit voffset(scale_offset).
__global__ __launch_bounds__(256) void morlet_wmma_kernel(
    const float* __restrict__ X,   // (TTOT, CHANS) f32
    const float* __restrict__ Aco, // (ETAS,) f32
    const float* __restrict__ Bco, // (ETAS,) f32
    float* __restrict__ out)       // (LOUT, ETAS, CHANS) f32
{
  const int lane  = threadIdx.x & 31;
  const int wave  = threadIdx.x >> 5;
  const int gwave = blockIdx.x * WAVES_PER_BLOCK + wave;
  const int l0    = gwave * ITER;
  const int mrow  = lane & 15;      // M (A) / N (B,D) index within tile
  const bool hi   = lane >= 16;
  const int hirow = hi ? 8 : 0;     // K-shift for A/B halves, M-shift for D rows

  // ---- A fragments (built once): ISA 16-bit 16x32 A layout ----
  // half h -> k = ((h<8)? h : h+8) + (hi? 8 : 0)
  v16h afrag[2];
#pragma unroll
  for (int mt = 0; mt < 2; ++mt) {
    const int e = mt * 16 + mrow;
    float av = 0.f, bv = 0.f;
    if (e < ETAS) { av = Aco[e]; bv = Bco[e]; }
#pragma unroll
    for (int h = 0; h < 16; ++h) {
      const int k = ((h < 8) ? h : h + 8) + hirow;
      float w = 0.f;
      if (e < ETAS && k < WLEN) {
        const float t = ((float)k * (1.0f / 12.0f) - 1.0f) * WTIME;
        w = __expf(-0.5f * av * av * t * t) * __cosf(TWO_PI * bv * t);
      }
      afrag[mt][h] = (half_t)w;
    }
  }

  // per-lane constant offset: K-base shift (loads) == M-row shift (stores)
  const int lane_off = hirow * CHANS + mrow;

  if (l0 + ITER - 1 <= TTOT - 32) {
    // ---- hot region (8188/8192 waves): no bounds checks anywhere ----
    int bidx0 = l0 * CHANS + lane_off;            // X element index
    int sidx0 = l0 * (ETAS * CHANS) + lane_off;   // out element index
    for (int it = 0; it < ITER; ++it) {
#pragma unroll
      for (int nt = 0; nt < 4; ++nt) {
        const int bidx = bidx0 + nt * 16;
        v16h bfrag;
#pragma unroll
        for (int h = 0; h < 16; ++h) {
          const int klo = (h < 8) ? h : h + 8;        // 0..7, 16..23
          bfrag[h] = (half_t)X[bidx + klo * CHANS];   // imm offsets off one base
        }

        v8f acc0 = {};
        acc0 = __builtin_amdgcn_wmma_f32_16x16x32_f16(false, afrag[0], false, bfrag,
                                                      (short)0, acc0, false, false);
        v8f acc1 = {};
        acc1 = __builtin_amdgcn_wmma_f32_16x16x32_f16(false, afrag[1], false, bfrag,
                                                      (short)0, acc1, false, false);

        const int sidx = sidx0 + nt * 16;
#pragma unroll
        for (int r = 0; r < 8; ++r)                   // e = r + 8*hi : all valid
          __builtin_nontemporal_store(acc0[r], out + sidx + r * CHANS);
        __builtin_nontemporal_store(acc1[0], out + sidx + 16 * CHANS); // e=16/24
        if (!hi) {                                    // e = 17..23, lo half only
#pragma unroll
          for (int r = 1; r < 8; ++r)
            __builtin_nontemporal_store(acc1[r], out + sidx + (16 + r) * CHANS);
        }
      }
      bidx0 += CHANS;
      sidx0 += ETAS * CHANS;
    }
  } else {
    // ---- tail waves: per-l guards + clamped rows ----
    for (int it = 0; it < ITER; ++it) {
      const int l = l0 + it;
      if (l >= LOUT) break;                           // wave-uniform
      const int sidx0 = l * (ETAS * CHANS) + lane_off;

#pragma unroll
      for (int nt = 0; nt < 4; ++nt) {
        const int cc = nt * 16 + mrow;
        v16h bfrag;
#pragma unroll
        for (int h = 0; h < 16; ++h) {
          const int klo = (h < 8) ? h : h + 8;
          int row = l + klo + hirow;
          row = (row < TTOT) ? row : (TTOT - 1);      // clamp; garbage killed by A
          bfrag[h] = (half_t)X[row * CHANS + cc];
        }

        v8f acc0 = {};
        acc0 = __builtin_amdgcn_wmma_f32_16x16x32_f16(false, afrag[0], false, bfrag,
                                                      (short)0, acc0, false, false);
        v8f acc1 = {};
        acc1 = __builtin_amdgcn_wmma_f32_16x16x32_f16(false, afrag[1], false, bfrag,
                                                      (short)0, acc1, false, false);

        const int sidx = sidx0 + nt * 16;
#pragma unroll
        for (int r = 0; r < 8; ++r)
          __builtin_nontemporal_store(acc0[r], out + sidx + r * CHANS);
        __builtin_nontemporal_store(acc1[0], out + sidx + 16 * CHANS);
        if (!hi) {
#pragma unroll
          for (int r = 1; r < 8; ++r)
            __builtin_nontemporal_store(acc1[r], out + sidx + (16 + r) * CHANS);
        }
      }
    }
  }
}

extern "C" void kernel_launch(void* const* d_in, const int* in_sizes, int n_in,
                              void* d_out, int out_size, void* d_ws, size_t ws_size,
                              hipStream_t stream) {
  const float* X   = (const float*)d_in[0]; // (65536, 64)
  const float* Aco = (const float*)d_in[1]; // (25, 1)
  const float* Bco = (const float*)d_in[2]; // (25, 1)
  float* out = (float*)d_out;               // (65512, 25, 64)

  const int total_waves = (LOUT + ITER - 1) / ITER;                         // 8189
  const int blocks = (total_waves + WAVES_PER_BLOCK - 1) / WAVES_PER_BLOCK; // 1024
  morlet_wmma_kernel<<<blocks, 32 * WAVES_PER_BLOCK, 0, stream>>>(X, Aco, Bco, out);
}
